// DeepseekMoE_35476429865913
// MI455X (gfx1250) — matrix operand order repaired and checked by reading, not compiled, not run on hardware
//
#include <hip/hip_runtime.h>
#include <hip/hip_bf16.h>

// ---- problem constants (from reference) ----
#define NTOK 65536
#define DDIM 256
#define HDIM 128
#define ODIM 256
#define NEXP 16
#define NEXP1 17          // 16 routed + shared expert as #16
#define TOPK 8
#define EPSV 1e-5f

#define MTT 4             // m-tiles (of 16 tokens) per workgroup = 64 tokens
#define TOKWG (MTT * 16)

// padded LDS row strides (halfs): stride%64dw==4 -> conflict-free row reads
#define XS_LD 264         // 256 + 8 pad  (528 B = 132 dw, 132%64=4)
#define H_LD  136         // 128 + 8 pad  (272 B =  68 dw,  68%64=4)

typedef __attribute__((ext_vector_type(16))) _Float16 v16h;
typedef __attribute__((ext_vector_type(8)))  _Float16 v8h;
typedef __attribute__((ext_vector_type(8)))  float    v8f;

// ------------------------------------------------------------------
// WMMA fragment helpers (CDNA5 wave32 layouts, 16x16x32 f16)
// ------------------------------------------------------------------
__device__ __forceinline__ v16h frag_a(const _Float16* p0, int ld, int k0,
                                       int hi, int r) {
  const _Float16* p = p0 + r * ld + k0 + hi * 8;
  v8h lo = *(const v8h*)(p);
  v8h hh = *(const v8h*)(p + 16);
  v16h o;
#pragma unroll
  for (int i = 0; i < 8; ++i) { o[i] = lo[i]; o[8 + i] = hh[i]; }
  return o;
}

__device__ __forceinline__ v16h frag_b(const _Float16* w, int ld, int n0, int k0,
                                       int hi, int r) {
  const _Float16* p = w + (size_t)(n0 + r) * ld + k0 + hi * 8;
  v8h lo = *(const v8h*)(p);
  v8h hh = *(const v8h*)(p + 16);
  v16h o;
#pragma unroll
  for (int i = 0; i < 8; ++i) { o[i] = lo[i]; o[8 + i] = hh[i]; }
  return o;
}

__device__ __forceinline__ v8f wmma32(v16h a, v16h b, v8f c) {
  return __builtin_amdgcn_wmma_f32_16x16x32_f16(false, a, false, b,
                                                (short)0, c, false, false);
}

// ------------------------------------------------------------------
// Kernel 1: f32 -> f16 cast
// ------------------------------------------------------------------
__global__ void cast_f32_to_f16(const float* __restrict__ in,
                                _Float16* __restrict__ out, int n) {
  int i = blockIdx.x * blockDim.x + threadIdx.x;
  int stride = gridDim.x * blockDim.x;
  for (; i < n; i += stride) out[i] = (_Float16)in[i];
}

// ------------------------------------------------------------------
// Kernel 2: fold eval-mode BatchNorm into next-layer weights.
//   BN(relu(z)) = relu(z)*a + c, a = g*rsqrt(rv+eps), c = bb - rm*a
//   W2' = W2 * a1 (along K), b2' = b2 + W2 @ c1 ; same for layer 3.
// ------------------------------------------------------------------
__global__ void prep_weights(
    const float* eW1, const float* eb1, const float* eg1, const float* ebb1,
    const float* erm1, const float* erv1,
    const float* eW2, const float* eb2, const float* eg2, const float* ebb2,
    const float* erm2, const float* erv2,
    const float* eW3, const float* eb3,
    const float* sW1, const float* sb1, const float* sg1, const float* sbb1,
    const float* srm1, const float* srv1,
    const float* sW2, const float* sb2, const float* sg2, const float* sbb2,
    const float* srm2, const float* srv2,
    const float* sW3, const float* sb3,
    _Float16* W1h, _Float16* W2h, _Float16* W3h,
    float* b1f, float* b2f, float* b3f) {
  __shared__ float a1[HDIM], c1[HDIM], a2[HDIM], c2[HDIM];
  const int e  = blockIdx.x;
  const int t  = threadIdx.x;
  const bool sh = (e == NEXP);
  const float* W1 = sh ? sW1 : eW1 + (size_t)e * HDIM * DDIM;
  const float* b1 = sh ? sb1 : eb1 + (size_t)e * HDIM;
  const float* g1 = sh ? sg1 : eg1 + (size_t)e * HDIM;
  const float* bb1 = sh ? sbb1 : ebb1 + (size_t)e * HDIM;
  const float* rm1 = sh ? srm1 : erm1 + (size_t)e * HDIM;
  const float* rv1 = sh ? srv1 : erv1 + (size_t)e * HDIM;
  const float* W2 = sh ? sW2 : eW2 + (size_t)e * HDIM * HDIM;
  const float* b2 = sh ? sb2 : eb2 + (size_t)e * HDIM;
  const float* g2 = sh ? sg2 : eg2 + (size_t)e * HDIM;
  const float* bb2 = sh ? sbb2 : ebb2 + (size_t)e * HDIM;
  const float* rm2 = sh ? srm2 : erm2 + (size_t)e * HDIM;
  const float* rv2 = sh ? srv2 : erv2 + (size_t)e * HDIM;
  const float* W3 = sh ? sW3 : eW3 + (size_t)e * ODIM * HDIM;
  const float* b3 = sh ? sb3 : eb3 + (size_t)e * ODIM;

  if (t < HDIM) {
    float a = g1[t] * rsqrtf(rv1[t] + EPSV);
    a1[t] = a;  c1[t] = bb1[t] - rm1[t] * a;
    float av = g2[t] * rsqrtf(rv2[t] + EPSV);
    a2[t] = av; c2[t] = bb2[t] - rm2[t] * av;
    b1f[(size_t)e * HDIM + t] = b1[t];
  }
  __syncthreads();

  for (int i = t; i < HDIM * DDIM; i += blockDim.x)
    W1h[(size_t)e * HDIM * DDIM + i] = (_Float16)W1[i];

  if (t < HDIM) {
    float acc = b2[t];
#pragma unroll 4
    for (int h = 0; h < HDIM; ++h) {
      float w = W2[t * HDIM + h];
      acc += w * c1[h];
      W2h[(size_t)e * HDIM * HDIM + t * HDIM + h] = (_Float16)(w * a1[h]);
    }
    b2f[(size_t)e * HDIM + t] = acc;
  }
  if (t < ODIM) {
    float acc = b3[t];
#pragma unroll 4
    for (int h = 0; h < HDIM; ++h) {
      float w = W3[t * HDIM + h];
      acc += w * c2[h];
      W3h[(size_t)e * ODIM * HDIM + t * HDIM + h] = (_Float16)(w * a2[h]);
    }
    b3f[(size_t)e * ODIM + t] = acc;
  }
}

// ------------------------------------------------------------------
// Kernel 3: fused gate + 17-expert MoE.
// Block = 256 thr = 8 waves, 64 tokens (4 m-tiles).
// Waves are N-parallel: wave w owns output column n0=w*16 (L1/L2),
// and columns {w, w+8}*16 in L3 -> each B fragment feeds 4 WMMAs.
// X tile staged once into LDS with async-to-LDS loads; reused by 17 experts.
// Register budget: 64 persistent acc + 32 layer acc + bounded fragment
// window (k-unroll 2) ~ 180 VGPRs -> no spills / no vgpr-msb traffic.
// ------------------------------------------------------------------
__global__ __launch_bounds__(256, 1) void moe_fused(
    const _Float16* __restrict__ X,      // [N, D] f16
    const _Float16* __restrict__ GW,     // [E, D] f16 gate weights
    const _Float16* __restrict__ W1h,    // [17, H, D]
    const _Float16* __restrict__ W2h,    // [17, H, H] (BN1-folded)
    const _Float16* __restrict__ W3h,    // [17, O, H] (BN2-folded)
    const float* __restrict__ b1f,       // [17, H]
    const float* __restrict__ b2f,       // [17, H]
    const float* __restrict__ b3f,       // [17, O]
    float* __restrict__ out) {           // [N, O]
  __shared__ __align__(16) _Float16 Xs[TOKWG * XS_LD];   // 33 KB, padded
  __shared__ __align__(16) _Float16 h1s[TOKWG * H_LD];   // 17 KB, padded
  __shared__ __align__(16) _Float16 h2s[TOKWG * H_LD];   // 17 KB, padded
  __shared__ float gts[TOKWG * 16];                      // logits
  __shared__ float wss[TOKWG * 18];                      // gate weights [m][e]

  const int tid  = threadIdx.x;
  const int lane = tid & 31;
  const int w    = tid >> 5;
  const int hi   = lane >> 4;
  const int r    = lane & 15;
  const int tok0 = blockIdx.x * TOKWG;

  // ---- stage X tile (64x256 f16) into LDS with async-to-LDS copies ----
  {
    const _Float16* Xg = X + (size_t)tok0 * DDIM;
#pragma unroll
    for (int i = 0; i < (TOKWG * 32) / 256; ++i) {   // 2048 chunks of 16B
      int ci  = tid + i * 256;
      int row = ci >> 5;
      int col = ci & 31;
      unsigned laddr = (unsigned)(uintptr_t)&Xs[row * XS_LD + col * 8];
      unsigned long long ga =
          (unsigned long long)(uintptr_t)(Xg + row * DDIM + col * 8);
      asm volatile("global_load_async_to_lds_b128 %0, %1, off"
                   :: "v"(laddr), "v"(ga) : "memory");
    }
    asm volatile("s_wait_asynccnt 0" ::: "memory");
  }
  __syncthreads();

  // ---------------- gate: logits tiles via WMMA (waves 0..3) -------------
  if (w < MTT) {
    v8f c = {};
    const _Float16* Xmt = &Xs[(w * 16) * XS_LD];
#pragma unroll 2
    for (int k = 0; k < DDIM / 32; ++k) {
      v16h a = frag_a(Xmt, XS_LD, k * 32, hi, r);
      v16h b = frag_b(GW, DDIM, 0, k * 32, hi, r);
      c = wmma32(a, b, c);
    }
#pragma unroll
    for (int j = 0; j < 8; ++j) gts[(w * 16 + j + 8 * hi) * 16 + r] = c[j];
  }
  __syncthreads();

  // softmax + top-8 + renormalize; one thread per token
  if (tid < TOKWG) {
    const int m = tid;
    float v[16];
#pragma unroll
    for (int e = 0; e < 16; ++e) v[e] = gts[m * 16 + e];
    float mx = v[0];
#pragma unroll
    for (int e = 1; e < 16; ++e) mx = fmaxf(mx, v[e]);
    float p[16];
#pragma unroll
    for (int e = 0; e < 16; ++e) p[e] = __expf(v[e] - mx);
    bool sel[16];
#pragma unroll
    for (int e = 0; e < 16; ++e) sel[e] = false;
    float sumTop = 0.f;
    for (int it = 0; it < TOPK; ++it) {
      int bi = 0; float bv = -1.f;
#pragma unroll
      for (int e = 0; e < 16; ++e)
        if (!sel[e] && p[e] > bv) { bv = p[e]; bi = e; }
      sel[bi] = true; sumTop += bv;
    }
    float inv = 1.f / (sumTop + 1e-20f);
#pragma unroll
    for (int e = 0; e < 16; ++e) wss[m * 18 + e] = sel[e] ? p[e] * inv : 0.f;
    wss[m * 18 + 16] = 1.0f;          // shared expert: weight 1
  }
  __syncthreads();

  // ---------------- expert loop ----------------
  float acc[2][MTT][8];               // persistent output accumulator
#pragma unroll
  for (int q = 0; q < 2; ++q)
#pragma unroll
    for (int mt = 0; mt < MTT; ++mt)
#pragma unroll
      for (int j = 0; j < 8; ++j) acc[q][mt][j] = 0.f;

  for (int e = 0; e < NEXP1; ++e) {
    const _Float16* W1 = W1h + (size_t)e * HDIM * DDIM;
    const _Float16* W2 = W2h + (size_t)e * HDIM * HDIM;
    const _Float16* W3 = W3h + (size_t)e * ODIM * HDIM;
    const float* B1 = b1f + (size_t)e * HDIM;
    const float* B2 = b2f + (size_t)e * HDIM;
    const float* B3 = b3f + (size_t)e * ODIM;
    if (e + 1 < NEXP1)
      __builtin_prefetch(W1h + (size_t)(e + 1) * HDIM * DDIM, 0, 0);

    // ---- layer 1: X[64,256] @ W1^T col (w*16..w*16+15); k-outer ----
    {
      v8f cc[MTT];
      float bv = B1[w * 16 + r];
#pragma unroll
      for (int mt = 0; mt < MTT; ++mt)
#pragma unroll
        for (int j = 0; j < 8; ++j) cc[mt][j] = bv;
#pragma unroll 2
      for (int k = 0; k < DDIM / 32; ++k) {
        v16h bf = frag_b(W1, DDIM, w * 16, k * 32, hi, r);
#pragma unroll
        for (int mt = 0; mt < MTT; ++mt) {
          v16h a = frag_a(&Xs[(mt * 16) * XS_LD], XS_LD, k * 32, hi, r);
          cc[mt] = wmma32(a, bf, cc[mt]);
        }
      }
#pragma unroll
      for (int mt = 0; mt < MTT; ++mt)
#pragma unroll
        for (int j = 0; j < 8; ++j)
          h1s[(mt * 16 + j + 8 * hi) * H_LD + w * 16 + r] =
              (_Float16)fmaxf(cc[mt][j], 0.f);
    }
    __syncthreads();

    // ---- layer 2: h1[64,128] @ W2'^T col; k-outer ----
    {
      v8f cc[MTT];
      float bv = B2[w * 16 + r];
#pragma unroll
      for (int mt = 0; mt < MTT; ++mt)
#pragma unroll
        for (int j = 0; j < 8; ++j) cc[mt][j] = bv;
#pragma unroll 2
      for (int k = 0; k < HDIM / 32; ++k) {
        v16h bf = frag_b(W2, HDIM, w * 16, k * 32, hi, r);
#pragma unroll
        for (int mt = 0; mt < MTT; ++mt) {
          v16h a = frag_a(&h1s[(mt * 16) * H_LD], H_LD, k * 32, hi, r);
          cc[mt] = wmma32(a, bf, cc[mt]);
        }
      }
#pragma unroll
      for (int mt = 0; mt < MTT; ++mt)
#pragma unroll
        for (int j = 0; j < 8; ++j)
          h2s[(mt * 16 + j + 8 * hi) * H_LD + w * 16 + r] =
              (_Float16)fmaxf(cc[mt][j], 0.f);
    }
    __syncthreads();

    // ---- layer 3: h2[64,128] @ W3'^T; wave owns cols {w, w+8}*16 ----
#pragma unroll
    for (int ntl = 0; ntl < 2; ++ntl) {
      const int n0 = (w + ntl * 8) * 16;
      v8f cc[MTT];
      float bv = B3[n0 + r];
#pragma unroll
      for (int mt = 0; mt < MTT; ++mt)
#pragma unroll
        for (int j = 0; j < 8; ++j) cc[mt][j] = bv;
#pragma unroll 2
      for (int k = 0; k < HDIM / 32; ++k) {
        v16h bf = frag_b(W3, HDIM, n0, k * 32, hi, r);
#pragma unroll
        for (int mt = 0; mt < MTT; ++mt) {
          v16h a = frag_a(&h2s[(mt * 16) * H_LD], H_LD, k * 32, hi, r);
          cc[mt] = wmma32(a, bf, cc[mt]);
        }
      }
#pragma unroll
      for (int mt = 0; mt < MTT; ++mt)
#pragma unroll
        for (int j = 0; j < 8; ++j) {
          float wv = wss[(mt * 16 + j + 8 * hi) * 18 + e];
          acc[ntl][mt][j] += wv * (1.f / (1.f + __expf(-cc[mt][j])));
        }
    }
    __syncthreads();
  }

  // ---------------- store [64, 256] output block ----------------
#pragma unroll
  for (int ntl = 0; ntl < 2; ++ntl)
#pragma unroll
    for (int mt = 0; mt < MTT; ++mt)
#pragma unroll
      for (int j = 0; j < 8; ++j)
        out[(size_t)(tok0 + mt * 16 + j + 8 * hi) * ODIM +
            (w + ntl * 8) * 16 + r] = acc[ntl][mt][j];
}

// ------------------------------------------------------------------
// launcher
// ------------------------------------------------------------------
extern "C" void kernel_launch(void* const* d_in, const int* in_sizes, int n_in,
                              void* d_out, int out_size, void* d_ws, size_t ws_size,
                              hipStream_t stream) {
  const float* combined = (const float*)d_in[0];
  const float* gate_w   = (const float*)d_in[1];
  const float* eW1 = (const float*)d_in[2];  const float* eb1 = (const float*)d_in[3];
  const float* eg1 = (const float*)d_in[4];  const float* ebb1 = (const float*)d_in[5];
  const float* erm1 = (const float*)d_in[6]; const float* erv1 = (const float*)d_in[7];
  const float* eW2 = (const float*)d_in[8];  const float* eb2 = (const float*)d_in[9];
  const float* eg2 = (const float*)d_in[10]; const float* ebb2 = (const float*)d_in[11];
  const float* erm2 = (const float*)d_in[12]; const float* erv2 = (const float*)d_in[13];
  const float* eW3 = (const float*)d_in[14]; const float* eb3 = (const float*)d_in[15];
  const float* sW1 = (const float*)d_in[16]; const float* sb1 = (const float*)d_in[17];
  const float* sg1 = (const float*)d_in[18]; const float* sbb1 = (const float*)d_in[19];
  const float* srm1 = (const float*)d_in[20]; const float* srv1 = (const float*)d_in[21];
  const float* sW2 = (const float*)d_in[22]; const float* sb2 = (const float*)d_in[23];
  const float* sg2 = (const float*)d_in[24]; const float* sbb2 = (const float*)d_in[25];
  const float* srm2 = (const float*)d_in[26]; const float* srv2 = (const float*)d_in[27];
  const float* sW3 = (const float*)d_in[28]; const float* sb3 = (const float*)d_in[29];
  float* out = (float*)d_out;

  // workspace layout (all offsets 256B aligned)
  char* ws = (char*)d_ws;
  size_t off = 0;
  _Float16* combinedH = (_Float16*)(ws + off); off += (size_t)NTOK * DDIM * 2;
  _Float16* gateWh    = (_Float16*)(ws + off); off += (size_t)NEXP * DDIM * 2;
  _Float16* W1h       = (_Float16*)(ws + off); off += (size_t)NEXP1 * HDIM * DDIM * 2;
  _Float16* W2h       = (_Float16*)(ws + off); off += (size_t)NEXP1 * HDIM * HDIM * 2;
  _Float16* W3h       = (_Float16*)(ws + off); off += (size_t)NEXP1 * ODIM * HDIM * 2;
  float* b1f          = (float*)(ws + off);    off += (size_t)NEXP1 * HDIM * 4;
  float* b2f          = (float*)(ws + off);    off += (size_t)NEXP1 * HDIM * 4;
  float* b3f          = (float*)(ws + off);    off += (size_t)NEXP1 * ODIM * 4;

  cast_f32_to_f16<<<1024, 256, 0, stream>>>(combined, combinedH, NTOK * DDIM);
  cast_f32_to_f16<<<16, 256, 0, stream>>>(gate_w, gateWh, NEXP * DDIM);

  prep_weights<<<NEXP1, 256, 0, stream>>>(
      eW1, eb1, eg1, ebb1, erm1, erv1,
      eW2, eb2, eg2, ebb2, erm2, erv2,
      eW3, eb3,
      sW1, sb1, sg1, sbb1, srm1, srv1,
      sW2, sb2, sg2, sbb2, srm2, srv2,
      sW3, sb3,
      W1h, W2h, W3h, b1f, b2f, b3f);

  moe_fused<<<NTOK / TOKWG, 256, 0, stream>>>(
      combinedH, gateWh, W1h, W2h, W3h, b1f, b2f, b3f, out);
}
